// DSConv_54597624267055
// MI455X (gfx1250) — matrix operand order repaired and checked by reading, not compiled, not run on hardware
//
#include <hip/hip_runtime.h>
#include <hip/hip_bf16.h>
#include <stdint.h>

// ---------------------------------------------------------------- types ----
typedef __attribute__((ext_vector_type(16))) __bf16         v16bf;
typedef __attribute__((ext_vector_type(8)))  float          v8f;
typedef __attribute__((ext_vector_type(4)))  unsigned int   v4u;
typedef __attribute__((ext_vector_type(4)))  int            v4i;
typedef __attribute__((ext_vector_type(8)))  int            v8i;
typedef __attribute__((ext_vector_type(4)))  unsigned short v4us;
typedef __attribute__((ext_vector_type(8)))  unsigned short v8us;
typedef __attribute__((ext_vector_type(16))) unsigned short v16us;

#define Bn    8
#define Cn    64
#define Hn    128
#define Wn    128
#define Kn    9
#define HWn   (Hn*Wn)        /* 16384   */
#define CHWn  (Cn*HWn)
#define Mn    (Bn*HWn)       /* 131072  */
#define KD    (Kn*Cn)        /* 576     */
#define NOFF  48             /* 36 offset channels padded to 3 N-tiles */
#define NMAIN 64
#define LPAD  40             /* LDS row stride in shorts: 80B = 16B-aligned,
                                r*20 mod 64 banks distinct for 16 rows */
#define EPSc  1e-5f

#if __has_builtin(__builtin_amdgcn_tensor_load_to_lds) && __has_builtin(__builtin_amdgcn_s_wait_tensorcnt)
#define USE_TDM 1
#else
#define USE_TDM 0
#endif

__device__ __forceinline__ unsigned short f2bf(float f) {
  unsigned u = __builtin_bit_cast(unsigned, f);
  u = u + 0x7FFFu + ((u >> 16) & 1u);          // round-to-nearest-even
  return (unsigned short)(u >> 16);
}

// Packed f32x2 -> bf16x2 (v_cvt_pk_bf16_f32 on gfx1250).
__device__ __forceinline__ unsigned pk2bf(float a, float b) {
  float2 f2; f2.x = a; f2.y = b;
  __hip_bfloat162 h2 = __float22bfloat162_rn(f2);
  unsigned u;
  __builtin_memcpy(&u, &h2, sizeof(u));        // bf16x2 bits
  return u;
}

// Two contiguous 16B LDS chunks -> one 16-lane bf16 fragment register block.
__device__ __forceinline__ v16bf frag2(const unsigned short* p0,
                                       const unsigned short* p1) {
  v8us lo = *(const v8us*)p0;                  // ds_load_b128
  v8us hi = *(const v8us*)p1;                  // ds_load_b128
  v16us c = __builtin_shufflevector(lo, hi, 0, 1, 2, 3, 4, 5, 6, 7,
                                    8, 9, 10, 11, 12, 13, 14, 15);
  return __builtin_bit_cast(v16bf, c);
}

// ------------------------------------------------------- TDM (D# packing) --
#if USE_TDM
// 2D tile (rows x cols of 2-byte elems, global row stride == cols) into LDS
// with hardware row padding: 16B inserted after every 64B (-> 80B LDS rows).
__device__ __forceinline__ void tdm_load_tile_b16_pad(const void* gsrc,
                                                      void* lds_dst,
                                                      int rows, int cols) {
  unsigned long long ga = (unsigned long long)(uintptr_t)gsrc;
  unsigned int       la = (unsigned int)(uintptr_t)lds_dst;
  v4u g0;
  g0.x = 1u;                                              // count=1, user mode
  g0.y = la;                                              // lds_addr
  g0.z = (unsigned int)(ga & 0xFFFFFFFFull);              // global_addr[31:0]
  g0.w = (unsigned int)((ga >> 32) & 0x1FFFFFFull) | (2u << 30); // [56:32]|type=2
  v8i g1;
  g1[0] = (int)((1u << 16)        // data_size = 1 (2B)
              | (1u << 20)        // pad_enable
              | (3u << 22)        // pad_interval: 16 DWORDs (64B)
              | (3u << 25));      // pad_amount:   4 DWORDs (16B)
  g1[1] = (int)((unsigned)cols << 16);      // tensor_dim0[15:0]  @ bits 63:48
  g1[2] = (int)((unsigned)rows << 16);      // tensor_dim1[15:0]  @ bits 95:80
  g1[3] = (int)((unsigned)cols << 16);      // tile_dim0          @ bits 127:112
  g1[4] = (int)(unsigned)rows;              // tile_dim1          @ bits 143:128
  g1[5] = (int)(unsigned)cols;              // tensor_dim0_stride @ bits 191:160
  g1[6] = 0;
  g1[7] = 0;
  v4i gz4 = {0, 0, 0, 0};
  v8i gz8 = {0, 0, 0, 0, 0, 0, 0, 0};
  __builtin_amdgcn_tensor_load_to_lds(g0, g1, gz4, gz4, gz8, 0); // TENSORcnt++
}
#endif

// ------------------------------------------------- workspace layout (bytes) -
static const size_t OFF_OFFRAW = 0;                         // 36*131072*4
static const size_t OFF_BNMEAN = 18874368;                  // 36 floats
static const size_t OFF_BNVAR  = OFF_BNMEAN + 256;
static const size_t OFF_CUMOFF = OFF_BNVAR + 256;           // 2*8*9*16384*4
static const size_t OFF_BOFF   = OFF_CUMOFF + 9437184;      // 18*48*32*2
static const size_t OFF_BMAIN  = OFF_BOFF + 55296;          // 2*18*64*32*2

// =================================================== 1) weight repack ======
// Transposed per-chunk layouts so LDS B fragments are contiguous per lane:
//   BoffT [chunk=0..17][col=0..47][kr=0..31]   (col<18: br0, col<36: br1)
//   BmainT[br][chunk=0..17][oc=0..63][kr=0..31]
// where global kidx = chunk*32 + kr; tap/k = kidx>>6, ic = kidx&63.
__global__ __launch_bounds__(256) void repack_weights(
    const float* __restrict__ ow0, const float* __restrict__ ow1,
    const float* __restrict__ cw0, const float* __restrict__ cw1,
    unsigned short* __restrict__ BoffT, unsigned short* __restrict__ BmainT) {
  int t = blockIdx.x * 256 + threadIdx.x;
  if (t < 18 * NOFF * 32) {
    int kkc = t / (NOFF * 32), r = t - kkc * (NOFF * 32);
    int col = r >> 5, kr = r & 31;
    int kidx = kkc * 32 + kr, tap = kidx >> 6, ic = kidx & 63;
    float v = 0.f;
    if (col < 18)      v = ow0[(col * Cn + ic) * 9 + tap];
    else if (col < 36) v = ow1[((col - 18) * Cn + ic) * 9 + tap];
    BoffT[t] = f2bf(v);
  }
  if (t < 2 * 18 * NMAIN * 32) {
    int br = t / (18 * NMAIN * 32), r = t - br * (18 * NMAIN * 32);
    int kkc = r / (NMAIN * 32), r2 = r - kkc * (NMAIN * 32);
    int oc = r2 >> 5, kr = r2 & 31;
    int kidx = kkc * 32 + kr, k = kidx >> 6, ic = kidx & 63;
    const float* w = br ? cw1 : cw0;               // (64,64,9,1)/(64,64,1,9)
    BmainT[t] = f2bf(w[(oc * Cn + ic) * 9 + k]);
  }
}

// ============================== 2) offset conv as WMMA implicit GEMM =======
// out: offraw[ch][m], ch 0..35, m = b*HW + h*W + w   (channel-major for BN)
// Halo handling is branchless: clamped address + 0/1 mask weight.
__global__ __launch_bounds__(128) void offset_gemm(
    const float* __restrict__ x, const unsigned short* __restrict__ BoffT,
    const float* __restrict__ ob0, const float* __restrict__ ob1,
    float* __restrict__ offraw) {
  __shared__ __align__(8) unsigned sIW[576][2];    // {clamped idx, mask bits}
  __shared__ __align__(16) unsigned short As2[64][LPAD];
  __shared__ __align__(16) unsigned short Bs2[NOFF][LPAD];
  const int m0 = blockIdx.x * 64;
  for (int e = threadIdx.x; e < 576; e += 128) {
    int ml = e / 9, tap = e - ml * 9, dy = tap / 3, dx = tap - dy * 3;
    int m = m0 + ml, b = m >> 14, pix = m & 16383, h = pix >> 7, w = pix & 127;
    int hh = h + dy - 1, ww = w + dx - 1;
    bool inb = (hh >= 0) & (hh < Hn) & (ww >= 0) & (ww < Wn);
    int hc = min(max(hh, 0), Hn - 1), wc = min(max(ww, 0), Wn - 1);
    sIW[e][0] = (unsigned)(b * CHWn + hc * Wn + wc);
    sIW[e][1] = __builtin_bit_cast(unsigned, inb ? 1.0f : 0.0f);
  }
  __syncthreads();
  const int lane = threadIdx.x & 31, wid = threadIdx.x >> 5;
  const int lm = lane & 15, khi = (lane >> 4) * 8, colb = lane & 15;
  const int klo = (lane < 16) ? 0 : 16;
  v8f acc[3] = {};
  for (int kk = 0; kk < 18; ++kk) {
    int tap = kk >> 1, c0 = (kk & 1) * 32;
    for (int q = threadIdx.x; q < 512; q += 128) {  // A tile: 4 ch per quad
      int ml = q >> 3, c4 = (q & 7) * 4, ci = ml * 9 + tap;
      unsigned idx = sIW[ci][0];                     // ds_load_b64
      float msk = __builtin_bit_cast(float, sIW[ci][1]);
      float s0 = msk * x[idx + (unsigned)(c0 + c4 + 0) * HWn];
      float s1 = msk * x[idx + (unsigned)(c0 + c4 + 1) * HWn];
      float s2 = msk * x[idx + (unsigned)(c0 + c4 + 2) * HWn];
      float s3 = msk * x[idx + (unsigned)(c0 + c4 + 3) * HWn];
      uint2 pk; pk.x = pk2bf(s0, s1); pk.y = pk2bf(s2, s3);
      *(uint2*)&As2[ml][c4] = pk;                   // ds_store_b64
    }
    const unsigned short* srcB = BoffT + kk * (NOFF * 32);
    for (int e = threadIdx.x; e < 384; e += 128) {  // B tile (transposed)
      int c = e >> 3, kq = (e & 7) * 4;
      *(v4us*)&Bs2[c][kq] = *(const v4us*)(srcB + c * 32 + kq);
    }
    __syncthreads();
    v16bf a = frag2(&As2[wid * 16 + lm][khi], &As2[wid * 16 + lm][16 + khi]);
#pragma unroll
    for (int t = 0; t < 3; ++t) {
      v16bf bf = frag2(&Bs2[t * 16 + colb][khi], &Bs2[t * 16 + colb][16 + khi]);
      acc[t] = __builtin_amdgcn_wmma_f32_16x16x32_bf16(
          false, a, false, bf, (short)0, acc[t], false, false);
    }
    __syncthreads();
  }
  const int rhi = (lane >> 4) * 8;
  const int mb = m0 + wid * 16 + rhi;               // 8 consecutive pixels
#pragma unroll
  for (int t = 0; t < 3; ++t) {
    int oc = t * 16 + colb;
    if (oc < 36) {
      float bias = (oc < 18) ? ob0[oc] : ob1[oc - 18];
      float4 f0 = {acc[t][0] + bias, acc[t][1] + bias,
                   acc[t][2] + bias, acc[t][3] + bias};
      float4 f1 = {acc[t][4] + bias, acc[t][5] + bias,
                   acc[t][6] + bias, acc[t][7] + bias};
      *(float4*)&offraw[(size_t)oc * Mn + mb]     = f0;   // b128 stores
      *(float4*)&offraw[(size_t)oc * Mn + mb + 4] = f1;
    }
  }
}

// =========================================== 3) batchnorm statistics =======
__global__ __launch_bounds__(256) void bn_stats(const float* __restrict__ offraw,
                                                float* __restrict__ mean,
                                                float* __restrict__ var) {
  __shared__ float rs[256], rq[256];
  const int c = blockIdx.x;
  float s = 0.f, q = 0.f;
  for (int i = threadIdx.x; i < Mn; i += 256) {
    float v = offraw[(size_t)c * Mn + i];
    s += v; q += v * v;
  }
  rs[threadIdx.x] = s; rq[threadIdx.x] = q;
  __syncthreads();
  for (int st = 128; st > 0; st >>= 1) {
    if (threadIdx.x < st) { rs[threadIdx.x] += rs[threadIdx.x + st];
                            rq[threadIdx.x] += rq[threadIdx.x + st]; }
    __syncthreads();
  }
  if (threadIdx.x == 0) {
    float m = rs[0] / (float)Mn;
    mean[c] = m;
    var[c]  = rq[0] / (float)Mn - m * m;
  }
}

// ================= 4) BN + tanh + cumulative snake offsets =================
// cumoff[br][b][k][pix]
__global__ __launch_bounds__(256) void make_coords(
    const float* __restrict__ offraw, const float* __restrict__ mean,
    const float* __restrict__ var,
    const float* __restrict__ g0, const float* __restrict__ b0,
    const float* __restrict__ g1, const float* __restrict__ b1,
    float* __restrict__ cumoff) {
  int t = blockIdx.x * 256 + threadIdx.x;
  if (t >= 2 * Mn) return;
  int br = t >> 17, m = t & (Mn - 1);
  int b = m >> 14, pix = m & 16383;
  int chbase = br ? 27 : 0;                    // br0: y_off ch 0..8, br1: x_off
  const float* gg = br ? g1 : g0;
  const float* gb = br ? b1 : b0;
  int pbase = br ? 9 : 0;
  float tv[9];
#pragma unroll
  for (int k = 0; k < 9; ++k) {
    int ch = chbase + k;
    float raw = offraw[(size_t)ch * Mn + m];
    float nv = (raw - mean[ch]) * rsqrtf(var[ch] + EPSc);
    tv[k] = tanhf(nv * gg[pbase + k] + gb[pbase + k]);
  }
  float cum[9];
  cum[4] = 0.f;
  float s = 0.f;
#pragma unroll
  for (int i = 3; i >= 0; --i) { s += tv[i]; cum[i] = s; }
  s = 0.f;
#pragma unroll
  for (int i = 5; i <= 8; ++i) { s += tv[i]; cum[i] = s; }
  size_t base = ((size_t)(br * Bn + b) * 9) * HWn;
#pragma unroll
  for (int k = 0; k < 9; ++k) cumoff[base + (size_t)k * HWn + pix] = cum[k];
}

// ====== 5) snake conv: bilinear-sampled implicit GEMM (WMMA bf16) ==========
// writes pre-groupnorm conv+bias output directly into d_out (NCHW).
__global__ __launch_bounds__(128) void snake_gemm(
    const float* __restrict__ x, const unsigned short* __restrict__ BmainT,
    const float* __restrict__ cumoff,
    const float* __restrict__ cb0, const float* __restrict__ cb1,
    float* __restrict__ out) {
  // 32B interleaved record per (pixel, tap): {idx00,idx01,idx10,idx11, w00..w11}
  __shared__ __align__(16) unsigned sIW[576][8];
  __shared__ __align__(16) unsigned short As2[64][LPAD];
  __shared__ __align__(16) unsigned short Bs2[NMAIN][LPAD];
  const int br = blockIdx.y;
  const int m0 = blockIdx.x * 64;
  for (int e = threadIdx.x; e < 576; e += 128) {  // coords shared by all 64 ch
    int ml = e / 9, k = e - ml * 9;
    int m = m0 + ml, b = m >> 14, pix = m & 16383, h = pix >> 7, w = pix & 127;
    float cum = cumoff[(((size_t)(br * Bn + b) * 9) + k) * HWn + pix];
    float yc, xc;
    if (br == 0) { yc = (float)h + cum; xc = (float)(w + (k - 4)); }
    else         { xc = (float)w + cum; yc = (float)(h + (k - 4)); }
    yc = fminf(fmaxf(yc, 0.f), 127.f);
    xc = fminf(fmaxf(xc, 0.f), 127.f);
    float y0f = floorf(yc), x0f = floorf(xc);
    int y0 = (int)y0f, x0i = (int)x0f;
    int y1 = min(y0 + 1, 127), x1 = min(x0i + 1, 127);
    float wy1 = yc - y0f, wy0 = 1.f - wy1, wx1 = xc - x0f, wx0 = 1.f - wx1;
    int base = b * CHWn;
    sIW[e][0] = (unsigned)(base + y0 * Wn + x0i);
    sIW[e][1] = (unsigned)(base + y0 * Wn + x1);
    sIW[e][2] = (unsigned)(base + y1 * Wn + x0i);
    sIW[e][3] = (unsigned)(base + y1 * Wn + x1);
    sIW[e][4] = __builtin_bit_cast(unsigned, wy0 * wx0);
    sIW[e][5] = __builtin_bit_cast(unsigned, wy0 * wx1);
    sIW[e][6] = __builtin_bit_cast(unsigned, wy1 * wx0);
    sIW[e][7] = __builtin_bit_cast(unsigned, wy1 * wx1);
  }
  __syncthreads();
  const int lane = threadIdx.x & 31, wid = threadIdx.x >> 5;
  const int lm = lane & 15, khi = (lane >> 4) * 8, colb = lane & 15;
  const int klo = (lane < 16) ? 0 : 16;
  const unsigned short* Bbr = BmainT + (size_t)br * 18 * NMAIN * 32;
  v8f acc[4] = {};
  for (int kk = 0; kk < 18; ++kk) {
    int k = kk >> 1, c0 = (kk & 1) * 32;
    const unsigned short* srcB = Bbr + kk * (NMAIN * 32);
#if USE_TDM
    if (wid == 0) {
      tdm_load_tile_b16_pad(srcB, &Bs2[0][0], NMAIN, 32);
      if (kk + 1 < 18)
        __builtin_prefetch(srcB + NMAIN * 32, 0, 1);
    }
#else
    for (int e = threadIdx.x; e < 512; e += 128) {
      int r = e >> 3, kq = (e & 7) * 4;
      *(v4us*)&Bs2[r][kq] = *(const v4us*)(srcB + r * 32 + kq);
    }
#endif
    for (int q = threadIdx.x; q < 512; q += 128) {  // bilinear A tile, 4ch/quad
      int ml = q >> 3, c4 = (q & 7) * 4, ci = ml * 9 + k;
      v4u   iv = *(const v4u*)&sIW[ci][0];          // ds_load_b128
      float4 wv = *(const float4*)&sIW[ci][4];      // ds_load_b128
      float sv[4];
#pragma unroll
      for (int j = 0; j < 4; ++j) {
        unsigned co = (unsigned)(c0 + c4 + j) * HWn;
        sv[j] = wv.x * x[iv.x + co] + wv.y * x[iv.y + co] +
                wv.z * x[iv.z + co] + wv.w * x[iv.w + co];
      }
      uint2 pk; pk.x = pk2bf(sv[0], sv[1]); pk.y = pk2bf(sv[2], sv[3]);
      *(uint2*)&As2[ml][c4] = pk;                   // ds_store_b64
    }
#if USE_TDM
    if (wid == 0) __builtin_amdgcn_s_wait_tensorcnt(0);
#endif
    __syncthreads();
    v16bf a = frag2(&As2[wid * 16 + lm][khi], &As2[wid * 16 + lm][16 + khi]);
#pragma unroll
    for (int t = 0; t < 4; ++t) {
      v16bf bf = frag2(&Bs2[t * 16 + colb][khi], &Bs2[t * 16 + colb][16 + khi]);
      acc[t] = __builtin_amdgcn_wmma_f32_16x16x32_bf16(
          false, a, false, bf, (short)0, acc[t], false, false);
    }
    __syncthreads();
  }
  const int rhi = (lane >> 4) * 8;
  const float* cb = br ? cb1 : cb0;
  const int mtile = m0 + wid * 16 + rhi;           // 8 consecutive pixels
  const int b = mtile >> 14, pixb = mtile & 16383;
#pragma unroll
  for (int t = 0; t < 4; ++t) {
    int oc = t * 16 + colb;
    float bias = cb[oc];
    size_t o = ((size_t)(b * 128 + br * 64 + oc)) * HWn + pixb;
    float4 f0 = {acc[t][0] + bias, acc[t][1] + bias,
                 acc[t][2] + bias, acc[t][3] + bias};
    float4 f1 = {acc[t][4] + bias, acc[t][5] + bias,
                 acc[t][6] + bias, acc[t][7] + bias};
    *(float4*)&out[o]     = f0;                    // global_store_b128
    *(float4*)&out[o + 4] = f1;
  }
}

// =========================== 6) groupnorm + relu in place ==================
__global__ __launch_bounds__(256) void gn_relu(
    float* __restrict__ out,
    const float* __restrict__ g0, const float* __restrict__ b0,
    const float* __restrict__ g1, const float* __restrict__ b1) {
  __shared__ float rs[256], rq[256];
  __shared__ float smean, sinv;
  int id = blockIdx.x;                 // 2 branches * 8 batch * 16 groups
  int br = id >> 7, rem = id & 127, b = rem >> 4, g = rem & 15;
  int chbase = br * 64 + g * 4;
  float s = 0.f, q = 0.f;
  for (int i = threadIdx.x; i < 4 * HWn; i += 256) {
    int cc = i >> 14, pix = i & 16383;
    float v = out[((size_t)(b * 128 + chbase + cc)) * HWn + pix];
    s += v; q += v * v;
  }
  rs[threadIdx.x] = s; rq[threadIdx.x] = q;
  __syncthreads();
  for (int st = 128; st > 0; st >>= 1) {
    if (threadIdx.x < st) { rs[threadIdx.x] += rs[threadIdx.x + st];
                            rq[threadIdx.x] += rq[threadIdx.x + st]; }
    __syncthreads();
  }
  if (threadIdx.x == 0) {
    float m = rs[0] / (float)(4 * HWn);
    smean = m;
    sinv  = rsqrtf(rq[0] / (float)(4 * HWn) - m * m + EPSc);
  }
  __syncthreads();
  const float* gg = br ? g1 : g0;
  const float* gb = br ? b1 : b0;
  float mn = smean, iv = sinv;
  for (int i = threadIdx.x; i < 4 * HWn; i += 256) {
    int cc = i >> 14, pix = i & 16383, lch = g * 4 + cc;
    size_t o = ((size_t)(b * 128 + chbase + cc)) * HWn + pix;
    float v = (out[o] - mn) * iv * gg[lch] + gb[lch];
    out[o] = v > 0.f ? v : 0.f;
  }
}

// ============================================================= launcher ====
extern "C" void kernel_launch(void* const* d_in, const int* in_sizes, int n_in,
                              void* d_out, int out_size, void* d_ws, size_t ws_size,
                              hipStream_t stream) {
  const float* x    = (const float*)d_in[0];
  const float* ow0  = (const float*)d_in[1];
  const float* ob0  = (const float*)d_in[2];
  const float* bg0  = (const float*)d_in[3];
  const float* bb0  = (const float*)d_in[4];
  const float* cw0  = (const float*)d_in[5];
  const float* cb0  = (const float*)d_in[6];
  const float* gg0  = (const float*)d_in[7];
  const float* gbb0 = (const float*)d_in[8];
  const float* ow1  = (const float*)d_in[9];
  const float* ob1  = (const float*)d_in[10];
  const float* bg1  = (const float*)d_in[11];
  const float* bb1  = (const float*)d_in[12];
  const float* cw1  = (const float*)d_in[13];
  const float* cb1  = (const float*)d_in[14];
  const float* gg1  = (const float*)d_in[15];
  const float* gbb1 = (const float*)d_in[16];
  float* out = (float*)d_out;
  char*  ws  = (char*)d_ws;

  float*          offraw = (float*)(ws + OFF_OFFRAW);
  float*          bnmean = (float*)(ws + OFF_BNMEAN);
  float*          bnvar  = (float*)(ws + OFF_BNVAR);
  float*          cumoff = (float*)(ws + OFF_CUMOFF);
  unsigned short* BoffT  = (unsigned short*)(ws + OFF_BOFF);
  unsigned short* BmainT = (unsigned short*)(ws + OFF_BMAIN);

  repack_weights<<<(2 * 18 * NMAIN * 32 + 255) / 256, 256, 0, stream>>>(
      ow0, ow1, cw0, cw1, BoffT, BmainT);
  offset_gemm<<<Mn / 64, 128, 0, stream>>>(x, BoffT, ob0, ob1, offraw);
  bn_stats<<<36, 256, 0, stream>>>(offraw, bnmean, bnvar);
  make_coords<<<(2 * Mn + 255) / 256, 256, 0, stream>>>(
      offraw, bnmean, bnvar, bg0, bb0, bg1, bb1, cumoff);
  snake_gemm<<<dim3(Mn / 64, 2), 128, 0, stream>>>(
      x, BmainT, cumoff, cb0, cb1, out);
  gn_relu<<<256, 256, 0, stream>>>(out, gg0, gbb0, gg1, gbb1);
}
